// CrossAttention_73856257622246
// MI455X (gfx1250) — matrix operand order repaired and checked by reading, not compiled, not run on hardware
//
#include <hip/hip_runtime.h>
#include <math.h>

typedef _Float16 v16h __attribute__((ext_vector_type(16)));
typedef _Float16 v8h  __attribute__((ext_vector_type(8)));
typedef float    v8f  __attribute__((ext_vector_type(8)));

#define WMMA_F32_F16(a, b, c) \
  __builtin_amdgcn_wmma_f32_16x16x32_f16(false, (a), false, (b), (short)0, (c), false, false)

constexpr int NB    = 4;     // batch
constexpr int QC    = 256;   // query channels
constexpr int TC    = 128;   // token_ch
constexpr int NTOK  = 256;   // tokens
constexpr int KVC   = 512;
constexpr int DH    = 32;
constexpr int NHEAD = 4;
constexpr int SVOL  = 16 * 32 * 32;   // 16384 spatial positions
constexpr int WAVES = 4;
constexpr int ROWS  = 16;             // spatial rows per wave

// A-fragment (16x32 f16) gather from an LDS row: one contiguous 16-half run
// at k = ka..ka+15 (ka = 0 or 16 by lane half).
static __device__ inline v16h lds_afrag(const _Float16* p) {
  v8h lo = *(const v8h*)(p);
  v8h hi = *(const v8h*)(p + 8);
  return __builtin_shufflevector(lo, hi, 0, 1, 2, 3, 4, 5, 6, 7,
                                 8, 9, 10, 11, 12, 13, 14, 15);
}

// ---------------------------------------------------------------------------
// Prep: f32 -> f16 weight conversion (run once per call, 640 KB total).
// ---------------------------------------------------------------------------
__global__ __launch_bounds__(256) void cvt_f16_kernel(const float* __restrict__ s,
                                                      _Float16* __restrict__ d,
                                                      int n) {
  int i = blockIdx.x * 256 + threadIdx.x;
  if (i < n) d[i] = (_Float16)s[i];
}

// ---------------------------------------------------------------------------
// Kernel 1: K/V projection. K pre-scaled by dh^-0.5.
// K layout  [b][h][n][32]   Vt layout [b][h][32][n]
// ---------------------------------------------------------------------------
__global__ __launch_bounds__(256) void kv_proj_kernel(
    const float* __restrict__ tokens, const float* __restrict__ kw,
    const float* __restrict__ vw, _Float16* __restrict__ Kb,
    _Float16* __restrict__ Vt) {
  __shared__ float ltok[2][KVC];
  int b  = blockIdx.x >> 7;              // 512 blocks: (b, n-pair)
  int n0 = (blockIdx.x & 127) * 2;
  for (int i = threadIdx.x; i < 2 * KVC; i += 256)
    ltok[i >> 9][i & (KVC - 1)] = tokens[((size_t)b * NTOK + n0) * KVC + i];
  __syncthreads();

  int t = threadIdx.x & (TC - 1);
  int n = n0 + (threadIdx.x >> 7);
  int h = t >> 5, d = t & 31;
  const float4* kr = (const float4*)(kw + (size_t)t * KVC);
  const float4* vr = (const float4*)(vw + (size_t)t * KVC);
  const float4* tk = (const float4*)ltok[threadIdx.x >> 7];
  float ak = 0.f, av = 0.f;
  #pragma unroll 4
  for (int c = 0; c < KVC / 4; ++c) {
    float4 x = tk[c], kk = kr[c], vv = vr[c];
    ak += x.x * kk.x + x.y * kk.y + x.z * kk.z + x.w * kk.w;
    av += x.x * vv.x + x.y * vv.y + x.z * vv.z + x.w * vv.w;
  }
  Kb[(((size_t)b * NHEAD + h) * NTOK + n) * DH + d] = (_Float16)(ak * 0.17677669529663689f);
  Vt[(((size_t)b * NHEAD + h) * DH + d) * NTOK + n] = (_Float16)av;
}

// ---------------------------------------------------------------------------
// Kernel 2: fused Q-proj -> attention -> proj -> gated residual.
// One wave owns 16 spatial rows; 4 waves per block; grid (SVOL/64, NB).
// ---------------------------------------------------------------------------
__global__ __launch_bounds__(128) void attn_kernel(
    const float* __restrict__ feat, const _Float16* __restrict__ qwh,
    const float* __restrict__ qb, const _Float16* __restrict__ Kb,
    const _Float16* __restrict__ Vt, const _Float16* __restrict__ projwh,
    const float* __restrict__ projb, const float* __restrict__ gate,
    float* __restrict__ outbuf) {
  __shared__ _Float16 ldsQ[WAVES][ROWS][TC];    // 16 KB
  __shared__ _Float16 ldsP[WAVES][ROWS][NTOK];  // 32 KB (reused for O)

  const int wave = threadIdx.x >> 5;
  const int lane = threadIdx.x & 31;
  const int b    = blockIdx.y;
  const int s0   = (blockIdx.x * WAVES + wave) * ROWS;
  const int row  = lane & 15;                 // A-matrix row (M)
  const int col  = lane & 15;                 // B/C column (N)
  const int ka   = (lane & 16) ? 16 : 0;      // A k-half offset
  const int kb16 = (lane & 16) ? 16 : 0;      // B k-half offset
  const int mb   = (lane & 16) ? 8 : 0;       // C row base

  // ---- GEMM1: Q[16x128] = feat_tile[16x256] * qw^T, + qb ----
  v16h af[8];
  #pragma unroll
  for (int kb = 0; kb < 8; ++kb) {
    const float* fp = feat + ((size_t)b * QC + kb * 32) * SVOL + s0 + row;
    #pragma unroll
    for (int i = 0; i < 16; ++i) {
      int k = i + ka;
      af[kb][i] = (_Float16)fp[(size_t)k * SVOL];
    }
  }
  #pragma unroll
  for (int nt = 0; nt < 8; ++nt) {
    v8f c = {};
    #pragma unroll
    for (int kb = 0; kb < 8; ++kb) {
      v16h bf = *(const v16h*)(qwh + (size_t)(nt * 16 + col) * QC + kb * 32 + kb16);
      c = WMMA_F32_F16(af[kb], bf, c);
    }
    float bias = qb[nt * 16 + col];
    #pragma unroll
    for (int j = 0; j < 8; ++j)
      ldsQ[wave][mb + j][nt * 16 + col] = (_Float16)(c[j] + bias);
  }

  // ---- attention per head; O accumulators stay in registers ----
  v8f oc[8];   // 8 column tiles of O[16x128]
  for (int h = 0; h < NHEAD; ++h) {
    const _Float16* Kh = Kb + (size_t)(b * NHEAD + h) * NTOK * DH;
    const _Float16* Vh = Vt + (size_t)(b * NHEAD + h) * DH * NTOK;
    // Q_h as A fragment (single K=32 block), from LDS
    v16h qa = lds_afrag(&ldsQ[wave][row][h * DH + ka]);
    // scores S[16x256] = Q_h * K_h^T  (K pre-scaled)
    v8f sc[16];
    #pragma unroll
    for (int nt = 0; nt < 16; ++nt) {
      v16h bf = *(const v16h*)(Kh + (size_t)(nt * 16 + col) * DH + kb16);
      v8f z = {};
      sc[nt] = WMMA_F32_F16(qa, bf, z);
    }
    // row softmax: each row of C lives in one 16-lane half
    #pragma unroll
    for (int j = 0; j < 8; ++j) {
      float mx = -3.0e38f;
      #pragma unroll
      for (int nt = 0; nt < 16; ++nt) mx = fmaxf(mx, sc[nt][j]);
      #pragma unroll
      for (int m = 1; m <= 8; m <<= 1) mx = fmaxf(mx, __shfl_xor(mx, m, 32));
      float sum = 0.f;
      #pragma unroll
      for (int nt = 0; nt < 16; ++nt) {
        float e = __expf(sc[nt][j] - mx);
        sc[nt][j] = e;
        sum += e;
      }
      #pragma unroll
      for (int m = 1; m <= 8; m <<= 1) sum += __shfl_xor(sum, m, 32);
      float inv = 1.f / sum;
      #pragma unroll
      for (int nt = 0; nt < 16; ++nt)
        ldsP[wave][mb + j][nt * 16 + col] = (_Float16)(sc[nt][j] * inv);
    }
    // O_h[16x32] = P[16x256] * V_h[256x32]
    #pragma unroll
    for (int dt = 0; dt < 2; ++dt) {
      v8f o = {};
      #pragma unroll
      for (int kb = 0; kb < 8; ++kb) {
        v16h pa = lds_afrag(&ldsP[wave][row][kb * 32 + ka]);
        v16h bf = *(const v16h*)(Vh + (size_t)(dt * 16 + col) * NTOK + kb * 32 + kb16);
        o = WMMA_F32_F16(pa, bf, o);
      }
      oc[h * 2 + dt] = o;
    }
  }

  // spill O (merged heads, 16x128) into the now-free P region for transposition
  #pragma unroll
  for (int t = 0; t < 8; ++t)
    #pragma unroll
    for (int j = 0; j < 8; ++j)
      ldsP[wave][mb + j][t * 16 + col] = (_Float16)oc[t][j];

  // ---- proj[16x256] = O[16x128] * projw^T, gated residual ----
  v16h oa[4];
  #pragma unroll
  for (int kb = 0; kb < 4; ++kb)
    oa[kb] = lds_afrag(&ldsP[wave][row][kb * 32 + ka]);
  float tg = tanhf(gate[0]);
  #pragma unroll
  for (int nt = 0; nt < 16; ++nt) {
    v8f c = {};
    #pragma unroll
    for (int kb = 0; kb < 4; ++kb) {
      v16h bf = *(const v16h*)(projwh + (size_t)(nt * 16 + col) * TC + kb * 32 + kb16);
      c = WMMA_F32_F16(oa[kb], bf, c);
    }
    float pb = projb[nt * 16 + col];
    int   ch = nt * 16 + col;
    // rows j=0..7 are 8 consecutive spatial positions -> two float4s
    size_t base = ((size_t)b * QC + ch) * SVOL + s0 + mb;
    float4 f0 = *(const float4*)(feat + base);
    float4 f1 = *(const float4*)(feat + base + 4);
    float4 o0, o1;
    o0.x = f0.x + tg * (c[0] + pb);
    o0.y = f0.y + tg * (c[1] + pb);
    o0.z = f0.z + tg * (c[2] + pb);
    o0.w = f0.w + tg * (c[3] + pb);
    o1.x = f1.x + tg * (c[4] + pb);
    o1.y = f1.y + tg * (c[5] + pb);
    o1.z = f1.z + tg * (c[6] + pb);
    o1.w = f1.w + tg * (c[7] + pb);
    *(float4*)(outbuf + base)     = o0;
    *(float4*)(outbuf + base + 4) = o1;
  }
}

// ---------------------------------------------------------------------------
// Kernel 3: FFN  (out -> ff1+silu -> ff2 -> += out), in place on outbuf.
// Rows are workgroup-private, so the read-modify-write is race-free.
// ---------------------------------------------------------------------------
__global__ __launch_bounds__(128) void ff_kernel(
    const _Float16* __restrict__ ff1wh, const float* __restrict__ ff1b,
    const _Float16* __restrict__ ff2wh, const float* __restrict__ ff2b,
    float* __restrict__ outbuf) {
  __shared__ _Float16 ldsH[WAVES][ROWS][2 * QC];  // 64 KB

  const int wave = threadIdx.x >> 5;
  const int lane = threadIdx.x & 31;
  const int b    = blockIdx.y;
  const int s0   = (blockIdx.x * WAVES + wave) * ROWS;
  const int row  = lane & 15, col = lane & 15;
  const int ka   = (lane & 16) ? 16 : 0;
  const int kb16 = (lane & 16) ? 16 : 0;
  const int mb   = (lane & 16) ? 8 : 0;

  // A fragments of the residual-in tile [16 x 256]
  v16h af[8];
  #pragma unroll
  for (int kb = 0; kb < 8; ++kb) {
    const float* fp = outbuf + ((size_t)b * QC + kb * 32) * SVOL + s0 + row;
    #pragma unroll
    for (int i = 0; i < 16; ++i) {
      int k = i + ka;
      af[kb][i] = (_Float16)fp[(size_t)k * SVOL];
    }
  }
  // hid[16x512] = tile * ff1w^T + ff1b, SiLU, to LDS
  for (int nt = 0; nt < 32; ++nt) {
    v8f c = {};
    #pragma unroll
    for (int kb = 0; kb < 8; ++kb) {
      v16h bf = *(const v16h*)(ff1wh + (size_t)(nt * 16 + col) * QC + kb * 32 + kb16);
      c = WMMA_F32_F16(af[kb], bf, c);
    }
    float bias = ff1b[nt * 16 + col];
    #pragma unroll
    for (int j = 0; j < 8; ++j) {
      float x = c[j] + bias;
      ldsH[wave][mb + j][nt * 16 + col] = (_Float16)(x / (1.f + __expf(-x)));
    }
  }
  // A fragments of hid (cached, reused over all 16 output tiles)
  v16h hf[16];
  #pragma unroll
  for (int kb = 0; kb < 16; ++kb)
    hf[kb] = lds_afrag(&ldsH[wave][row][kb * 32 + ka]);
  // final[16x256] = hid * ff2w^T + ff2b + residual
  for (int nt = 0; nt < 16; ++nt) {
    v8f c = {};
    #pragma unroll
    for (int kb = 0; kb < 16; ++kb) {
      v16h bf = *(const v16h*)(ff2wh + (size_t)(nt * 16 + col) * (2 * QC) + kb * 32 + kb16);
      c = WMMA_F32_F16(hf[kb], bf, c);
    }
    float bias = ff2b[nt * 16 + col];
    int   ch   = nt * 16 + col;
    size_t base = ((size_t)b * QC + ch) * SVOL + s0 + mb;
    float4 r0 = *(const float4*)(outbuf + base);
    float4 r1 = *(const float4*)(outbuf + base + 4);
    r0.x += c[0] + bias;
    r0.y += c[1] + bias;
    r0.z += c[2] + bias;
    r0.w += c[3] + bias;
    r1.x += c[4] + bias;
    r1.y += c[5] + bias;
    r1.z += c[6] + bias;
    r1.w += c[7] + bias;
    *(float4*)(outbuf + base)     = r0;
    *(float4*)(outbuf + base + 4) = r1;
  }
}

// ---------------------------------------------------------------------------
extern "C" void kernel_launch(void* const* d_in, const int* in_sizes, int n_in,
                              void* d_out, int out_size, void* d_ws, size_t ws_size,
                              hipStream_t stream) {
  const float* feat   = (const float*)d_in[0];
  const float* tokens = (const float*)d_in[1];
  const float* qw     = (const float*)d_in[2];
  const float* qb     = (const float*)d_in[3];
  const float* kw     = (const float*)d_in[4];
  const float* vw     = (const float*)d_in[5];
  const float* projw  = (const float*)d_in[6];
  const float* projb  = (const float*)d_in[7];
  const float* ff1w   = (const float*)d_in[8];
  const float* ff1b   = (const float*)d_in[9];
  const float* ff2w   = (const float*)d_in[10];
  const float* ff2b   = (const float*)d_in[11];
  const float* gate   = (const float*)d_in[12];
  float* outbuf = (float*)d_out;

  // workspace layout (halfs)
  _Float16* Kb     = (_Float16*)d_ws;                        // 131072 h
  _Float16* Vt     = Kb + (size_t)NB * NHEAD * NTOK * DH;    // 131072 h
  _Float16* qwh    = Vt + (size_t)NB * NHEAD * NTOK * DH;    // 32768 h
  _Float16* projwh = qwh + TC * QC;                          // 32768 h
  _Float16* ff1wh  = projwh + QC * TC;                       // 131072 h
  _Float16* ff2wh  = ff1wh + 2 * QC * QC;                    // 131072 h

  // weight conversion (tiny, once per call)
  cvt_f16_kernel<<<dim3((TC * QC + 255) / 256), dim3(256), 0, stream>>>(qw, qwh, TC * QC);
  cvt_f16_kernel<<<dim3((QC * TC + 255) / 256), dim3(256), 0, stream>>>(projw, projwh, QC * TC);
  cvt_f16_kernel<<<dim3((2 * QC * QC + 255) / 256), dim3(256), 0, stream>>>(ff1w, ff1wh, 2 * QC * QC);
  cvt_f16_kernel<<<dim3((2 * QC * QC + 255) / 256), dim3(256), 0, stream>>>(ff2w, ff2wh, 2 * QC * QC);

  kv_proj_kernel<<<dim3(NB * NTOK / 2), dim3(256), 0, stream>>>(tokens, kw, vw, Kb, Vt);
  attn_kernel<<<dim3(SVOL / (WAVES * ROWS), NB), dim3(WAVES * 32), 0, stream>>>(
      feat, qwh, qb, Kb, Vt, projwh, projb, gate, outbuf);
  ff_kernel<<<dim3(SVOL / (WAVES * ROWS), NB), dim3(WAVES * 32), 0, stream>>>(
      ff1wh, ff1b, ff2wh, ff2b, outbuf);
}
